// LlamaAttention_28372553957510
// MI455X (gfx1250) — compile-verified, matrix operand order checked by reading, not run on hardware
//
#include <hip/hip_runtime.h>

typedef __attribute__((ext_vector_type(16))) _Float16 v16h;
typedef __attribute__((ext_vector_type(8)))  _Float16 v8h;
typedef __attribute__((ext_vector_type(8)))  float    v8f;

#define B_   2
#define S_   2048
#define H_   2048
#define NH_  32
#define NKV_ 8
#define D_   64
#define GROUPS_ (NH_ / NKV_)
#define ND_  (NH_ * D_)     // 2048
#define KVD_ (NKV_ * D_)    // 512
#define SCALE_ 0.125f       // D^-0.5

union Frag {
    v16h v;
    struct { v8h lo; v8h hi; } p;
};

// ---------------------------------------------------------------------------
// fp32 -> fp16 downcast (x and the four weight matrices)
// ---------------------------------------------------------------------------
__global__ void cvt_f32_f16(const float* __restrict__ src,
                            _Float16* __restrict__ dst, int n) {
    int i = blockIdx.x * blockDim.x + threadIdx.x;
    if (i < n) dst[i] = (_Float16)src[i];
}

// ---------------------------------------------------------------------------
// C[M,N] = A[M,K] * W[N,K]^T   (both f16 row-major along K, f32 accumulate)
// One wave32 per 32(M) x 64(N) output tile: 2 A-fragments x 4 B-fragments
// -> 8 v_wmma_f32_16x16x32_f16 per K=32 step (~21 FLOP/byte of fragment
// traffic vs 8 for a 16x16 tile). A-fragment per ISA 7.12.2: lane%16 = row,
// lane/16 picks the K-half; two b128 loads per fragment. B-fragment =
// A-style load of W (since B = W^T and layouts are transpose-symmetric).
// ---------------------------------------------------------------------------
__global__ __launch_bounds__(256)
void gemm_nt_wmma(const _Float16* __restrict__ A,
                  const _Float16* __restrict__ W,
                  float* __restrict__ C, int M, int N, int K) {
    int lane = threadIdx.x & 31;
    int wid  = (int)((blockIdx.x * blockDim.x + threadIdx.x) >> 5);
    int tilesN = N >> 6;              // 64-wide N tiles
    int tilesM = M >> 5;              // 32-tall M tiles
    int mt = wid / tilesN;
    int nt = wid % tilesN;
    if (mt >= tilesM) return;

    int l16 = lane & 15;
    int g   = lane >> 4;              // K-half selector

    const _Float16* arow0 = A + (size_t)(mt * 32 + l16) * K + g * 8;
    const _Float16* arow1 = arow0 + (size_t)16 * K;
    const _Float16* wrow[4];
    #pragma unroll
    for (int j = 0; j < 4; ++j)
        wrow[j] = W + (size_t)(nt * 64 + j * 16 + l16) * K + g * 8;

    v8f acc[2][4] = {};
    for (int k = 0; k < K; k += 32) {
        Frag a0, a1;
        a0.p.lo = *(const v8h*)(arow0 + k);
        a0.p.hi = *(const v8h*)(arow0 + k + 16);
        a1.p.lo = *(const v8h*)(arow1 + k);
        a1.p.hi = *(const v8h*)(arow1 + k + 16);
        #pragma unroll
        for (int j = 0; j < 4; ++j) {
            Frag b;
            b.p.lo = *(const v8h*)(wrow[j] + k);
            b.p.hi = *(const v8h*)(wrow[j] + k + 16);
            acc[0][j] = __builtin_amdgcn_wmma_f32_16x16x32_f16(
                            false, a0.v, false, b.v, (short)0, acc[0][j], false, false);
            acc[1][j] = __builtin_amdgcn_wmma_f32_16x16x32_f16(
                            false, a1.v, false, b.v, (short)0, acc[1][j], false, false);
        }
    }

    // C-layout: VGPR r -> row = base + g*8 + r, col = base + l16
    #pragma unroll
    for (int i = 0; i < 2; ++i) {
        int m0 = mt * 32 + i * 16 + g * 8;
        #pragma unroll
        for (int j = 0; j < 4; ++j) {
            int n = nt * 64 + j * 16 + l16;
            #pragma unroll
            for (int r = 0; r < 8; ++r)
                C[(size_t)(m0 + r) * N + n] = acc[i][j][r];
        }
    }
}

// ---------------------------------------------------------------------------
// RoPE on q,k (f32 in) + repack to f16 attention layouts:
//   qh: (B, NH, S, D)   kh: (B, NKV, S, D)   vt: (B, NKV, D, S) transposed
// ---------------------------------------------------------------------------
__global__ void rope_pack(const float* __restrict__ qf,
                          const float* __restrict__ kf,
                          const float* __restrict__ vf,
                          const float* __restrict__ cosp,
                          const float* __restrict__ sinp,
                          _Float16* __restrict__ qh,
                          _Float16* __restrict__ kh,
                          _Float16* __restrict__ vt) {
    const int HTOT = NH_ + 2 * NKV_;
    int idx = blockIdx.x * blockDim.x + threadIdx.x;
    if (idx >= B_ * S_ * HTOT * D_) return;
    int d  = idx % D_;
    int t  = idx / D_;
    int hh = t % HTOT;  t /= HTOT;
    int s  = t % S_;
    int b  = t / S_;

    size_t cs = (size_t)(b * S_ + s) * D_ + d;
    float c  = cosp[cs];
    float sn = sinp[cs];

    if (hh < NH_) {
        size_t base = (size_t)(b * S_ + s) * ND_ + hh * D_;
        float x  = qf[base + d];
        float xo = (d < D_ / 2) ? -qf[base + d + D_ / 2] : qf[base + d - D_ / 2];
        qh[((size_t)(b * NH_ + hh) * S_ + s) * D_ + d] = (_Float16)(x * c + xo * sn);
    } else if (hh < NH_ + NKV_) {
        int hk = hh - NH_;
        size_t base = (size_t)(b * S_ + s) * KVD_ + hk * D_;
        float x  = kf[base + d];
        float xo = (d < D_ / 2) ? -kf[base + d + D_ / 2] : kf[base + d - D_ / 2];
        kh[((size_t)(b * NKV_ + hk) * S_ + s) * D_ + d] = (_Float16)(x * c + xo * sn);
    } else {
        int hk = hh - NH_ - NKV_;
        float x = vf[(size_t)(b * S_ + s) * KVD_ + hk * D_ + d];
        vt[((size_t)(b * NKV_ + hk) * D_ + d) * S_ + s] = (_Float16)x;
    }
}

// ---------------------------------------------------------------------------
// Flash attention: one wave per 16-row q-tile per (b, h).
// Per 32-key tile: 4 WMMAs (Q.K^T), online softmax (half-wave shuffles),
// P transposed C-layout -> A-layout through per-wave LDS, 4 WMMAs (P.V).
// Causal mask handled analytically; emits f16 (B, S, NH*D) for the O-proj.
// ---------------------------------------------------------------------------
__global__ __launch_bounds__(128)
void flash_attn(const _Float16* __restrict__ qh,
                const _Float16* __restrict__ kh,
                const _Float16* __restrict__ vt,
                _Float16* __restrict__ oh) {
    __shared__ __align__(16) _Float16 pbuf[4][16 * 32];

    int lane = threadIdx.x & 31;
    int w    = threadIdx.x >> 5;
    int l16  = lane & 15;
    int g    = lane >> 4;

    int bh = blockIdx.y;
    int b  = bh / NH_;
    int h  = bh % NH_;
    int hk = h / GROUPS_;
    int qbase = (blockIdx.x * 4 + w) * 16;

    const _Float16* Q  = qh + (size_t)(b * NH_  + h ) * S_ * D_;
    const _Float16* Kp = kh + (size_t)(b * NKV_ + hk) * S_ * D_;
    const _Float16* Vt = vt + (size_t)(b * NKV_ + hk) * D_ * S_;

    // Q fragments for d = 0..31 and 32..63
    Frag qa0, qa1;
    {
        const _Float16* qr = Q + (size_t)(qbase + l16) * D_ + g * 8;
        qa0.p.lo = *(const v8h*)(qr);
        qa0.p.hi = *(const v8h*)(qr + 16);
        qa1.p.lo = *(const v8h*)(qr + 32);
        qa1.p.hi = *(const v8h*)(qr + 48);
    }

    float m[8], l[8];
    v8f oacc[4] = {};
    #pragma unroll
    for (int r = 0; r < 8; ++r) { m[r] = -1e30f; l[r] = 0.0f; }

    int kend = qbase + 16;
    for (int kb = 0; kb < kend; kb += 32) {
        if (kb + 32 < kend)
            __builtin_prefetch(Kp + (size_t)(kb + 32 + l16) * D_, 0, 0);

        // ---- scores: S = Q (16x64) . K^T, two 16-key subtiles ----
        v8f sacc[2];
        #pragma unroll
        for (int sub = 0; sub < 2; ++sub) {
            const _Float16* kr = Kp + (size_t)(kb + sub * 16 + l16) * D_ + g * 8;
            Frag k0, k1;
            k0.p.lo = *(const v8h*)(kr);
            k0.p.hi = *(const v8h*)(kr + 16);
            k1.p.lo = *(const v8h*)(kr + 32);
            k1.p.hi = *(const v8h*)(kr + 48);
            v8f s = {};
            s = __builtin_amdgcn_wmma_f32_16x16x32_f16(
                    false, qa0.v, false, k0.v, (short)0, s, false, false);
            s = __builtin_amdgcn_wmma_f32_16x16x32_f16(
                    false, qa1.v, false, k1.v, (short)0, s, false, false);
            sacc[sub] = s;
        }

        // ---- scale + causal mask (C-layout: row = qbase + g*8 + r) ----
        #pragma unroll
        for (int sub = 0; sub < 2; ++sub) {
            #pragma unroll
            for (int r = 0; r < 8; ++r) {
                float sv  = sacc[sub][r] * SCALE_;
                int   row = qbase + g * 8 + r;
                int   key = kb + sub * 16 + l16;
                sacc[sub][r] = (key > row) ? -1e30f : sv;
            }
        }

        // ---- online softmax; rows live across 16 lanes of a half-wave ----
        #pragma unroll
        for (int r = 0; r < 8; ++r) {
            float rmax = fmaxf(sacc[0][r], sacc[1][r]);
            #pragma unroll
            for (int sh = 8; sh >= 1; sh >>= 1)
                rmax = fmaxf(rmax, __shfl_xor(rmax, sh, 32));
            float mnew = fmaxf(m[r], rmax);
            float fac  = __expf(m[r] - mnew);
            float p0   = __expf(sacc[0][r] - mnew);
            float p1   = __expf(sacc[1][r] - mnew);
            sacc[0][r] = p0;
            sacc[1][r] = p1;
            float rsum = p0 + p1;
            #pragma unroll
            for (int sh = 8; sh >= 1; sh >>= 1)
                rsum += __shfl_xor(rsum, sh, 32);
            l[r] = l[r] * fac + rsum;
            m[r] = mnew;
            #pragma unroll
            for (int t = 0; t < 4; ++t) oacc[t][r] *= fac;
        }

        // ---- transpose P: C-layout -> A-layout via per-wave LDS ----
        #pragma unroll
        for (int sub = 0; sub < 2; ++sub) {
            #pragma unroll
            for (int r = 0; r < 8; ++r)
                pbuf[w][(g * 8 + r) * 32 + sub * 16 + l16] = (_Float16)sacc[sub][r];
        }
        Frag pa;   // LDS ops are in-order per wave; compiler inserts dscnt waits
        pa.p.lo = *(const v8h*)&pbuf[w][l16 * 32 + g * 8];
        pa.p.hi = *(const v8h*)&pbuf[w][l16 * 32 + g * 8 + 16];

        // ---- out += P (16x32) . V (32 x 64): 4 WMMAs over d-tiles ----
        #pragma unroll
        for (int t = 0; t < 4; ++t) {
            const _Float16* vr = Vt + (size_t)(t * 16 + l16) * S_ + kb + g * 8;
            Frag vb;
            vb.p.lo = *(const v8h*)(vr);
            vb.p.hi = *(const v8h*)(vr + 16);
            oacc[t] = __builtin_amdgcn_wmma_f32_16x16x32_f16(
                          false, pa.v, false, vb.v, (short)0, oacc[t], false, false);
        }
    }

    // ---- epilogue: normalize and emit f16 (B, S, NH*D) ----
    #pragma unroll
    for (int t = 0; t < 4; ++t) {
        #pragma unroll
        for (int r = 0; r < 8; ++r) {
            int row = qbase + g * 8 + r;
            float val = oacc[t][r] / l[r];
            oh[(size_t)(b * S_ + row) * ND_ + h * D_ + t * 16 + l16] = (_Float16)val;
        }
    }
}

// ---------------------------------------------------------------------------
// Host-side orchestration
// ---------------------------------------------------------------------------
extern "C" void kernel_launch(void* const* d_in, const int* in_sizes, int n_in,
                              void* d_out, int out_size, void* d_ws, size_t ws_size,
                              hipStream_t stream) {
    const float* x    = (const float*)d_in[0];
    const float* cosp = (const float*)d_in[1];
    const float* sinp = (const float*)d_in[2];
    // d_in[3] = causal mask: handled analytically in flash_attn
    const float* wq   = (const float*)d_in[4];
    const float* wk   = (const float*)d_in[5];
    const float* wv   = (const float*)d_in[6];
    const float* wo   = (const float*)d_in[7];
    float* out = (float*)d_out;

    char* ws = (char*)d_ws;
    size_t off = 0;
    auto alloc = [&](size_t bytes) -> char* {
        char* p = ws + off;
        off += (bytes + 255) & ~(size_t)255;
        return p;
    };
    const size_t MS = (size_t)B_ * S_;   // 4096 token rows

    _Float16* xh  = (_Float16*)alloc(MS * H_ * 2);
    _Float16* wqh = (_Float16*)alloc((size_t)ND_  * H_ * 2);
    _Float16* wkh = (_Float16*)alloc((size_t)KVD_ * H_ * 2);
    _Float16* wvh = (_Float16*)alloc((size_t)KVD_ * H_ * 2);
    _Float16* woh = (_Float16*)alloc((size_t)H_ * ND_ * 2);
    float*    qf  = (float*)   alloc(MS * ND_  * 4);
    float*    kf  = (float*)   alloc(MS * KVD_ * 4);
    float*    vf  = (float*)   alloc(MS * KVD_ * 4);
    _Float16* qhh = (_Float16*)alloc(MS * ND_  * 2);
    _Float16* khh = (_Float16*)alloc(MS * KVD_ * 2);
    _Float16* vth = (_Float16*)alloc(MS * KVD_ * 2);
    _Float16* ohh = (_Float16*)alloc(MS * ND_  * 2);

    auto cvt = [&](const float* s, _Float16* dp, size_t n) {
        cvt_f32_f16<<<(unsigned)((n + 255) / 256), 256, 0, stream>>>(s, dp, (int)n);
    };
    cvt(x,  xh,  MS * H_);
    cvt(wq, wqh, (size_t)ND_  * H_);
    cvt(wk, wkh, (size_t)KVD_ * H_);
    cvt(wv, wvh, (size_t)KVD_ * H_);
    cvt(wo, woh, (size_t)H_ * ND_);

    auto gemm = [&](const _Float16* A, const _Float16* W, float* C,
                    int M, int N, int K) {
        size_t waves = (size_t)(M / 32) * (size_t)(N / 64);
        unsigned blocks = (unsigned)((waves + 7) / 8);   // 8 waves / 256-thr block
        gemm_nt_wmma<<<blocks, 256, 0, stream>>>(A, W, C, M, N, K);
    };
    gemm(xh, wqh, qf, (int)MS, ND_,  H_);
    gemm(xh, wkh, kf, (int)MS, KVD_, H_);
    gemm(xh, wvh, vf, (int)MS, KVD_, H_);

    {
        int total = B_ * S_ * (NH_ + 2 * NKV_) * D_;
        rope_pack<<<(total + 255) / 256, 256, 0, stream>>>(
            qf, kf, vf, cosp, sinp, qhh, khh, vth);
    }

    dim3 agrid(S_ / 16 / 4, B_ * NH_);
    flash_attn<<<agrid, 128, 0, stream>>>(qhh, khh, vth, ohh);

    gemm(ohh, woh, out, (int)MS, H_, ND_);

    (void)in_sizes; (void)n_in; (void)out_size; (void)ws_size;
}
